// Resnet18_graph_43516608643446
// MI455X (gfx1250) — matrix operand order repaired and checked by reading, not compiled, not run on hardware
//
#include <hip/hip_runtime.h>
#include <math.h>

typedef __attribute__((ext_vector_type(16))) __bf16 v16bf;
typedef __attribute__((ext_vector_type(8)))  float  v8f;

#define BN_EPS 1e-5f

#define SC_PLAIN 0
#define SC_BNELU 1
#define SC_POOL  2
#define SC_DISP  3

__device__ __forceinline__ unsigned short f2bf(float f) {
  unsigned int u = __float_as_uint(f);
  u += 0x7FFFu + ((u >> 16) & 1u);           // round-to-nearest-even
  return (unsigned short)(u >> 16);
}

union FragBF { unsigned short u[16]; uint4 q[2]; v16bf v; };

// ---------------------------------------------------------------------------
// Selection-conv (3x3 graph conv) as gathered GEMM on the WMMA pipe.
// Block = 128 threads = 4 wave32. Per K-chunk the block shares one gathered
// A tile of 32 nodes x 32 channels (bf16, LDS); wave w owns cout tile
// [blockIdx.y*64 + w*16, +16) and keeps TWO 16x16 f32 accumulators
// (rows 0-15 and 16-31 of the A tile), so each LDS round-trip issues two
// v_wmma_f32_16x16x32_bf16 sharing one B fragment.
// LDS is laid out so每 fragment is loaded with two ds_load_b128:
//   A row-major [32][40] (80B stride, 16B aligned): lane fragment = two
//     contiguous 16B runs; B stored transposed [16 cols][40] so a lane's
//     fragment is likewise two 16B runs (same k order as A).
// ---------------------------------------------------------------------------
__global__ void __launch_bounds__(128) sconv_wmma_kernel(
    const float* __restrict__ x, const int* __restrict__ nbr,
    const float* __restrict__ W, const float* __restrict__ bias,
    const float* __restrict__ g, const float* __restrict__ be,
    const float* __restrict__ mn, const float* __restrict__ vr,
    const int* __restrict__ cluster, float* __restrict__ out,
    int N, int Cin, int Cout, int numSel, int mode)
{
  __shared__ __align__(16) unsigned short As[32][40];     // 32 nodes x 32 k
  __shared__ __align__(16) unsigned short Bt[4][16][40];  // per-wave, [col][k]

  const int tid  = threadIdx.x;
  const int lane = tid & 31;
  const int wv   = tid >> 5;
  const int m0   = blockIdx.x * 32;
  const int coutBase = blockIdx.y * 64 + wv * 16;

  v8f acc0 = {0.f, 0.f, 0.f, 0.f, 0.f, 0.f, 0.f, 0.f};
  v8f acc1 = {0.f, 0.f, 0.f, 0.f, 0.f, 0.f, 0.f, 0.f};

  const int mrow = lane & 15;
  const int half = lane >> 4;

  for (int s = 0; s < numSel; ++s) {
    for (int k0 = 0; k0 < Cin; k0 += 32) {
      // cooperative A: 32 nodes x 32 channels, gathered through nbr map
      for (int e = tid; e < 1024; e += 128) {
        int m    = e >> 5;
        int kk   = e & 31;
        int cin  = k0 + kk;
        int node = m0 + m;
        int src  = nbr ? nbr[node * 9 + s] : node;
        float v = 0.f;
        if (src >= 0 && cin < Cin) v = x[(size_t)src * Cin + cin];
        As[m][kk] = f2bf(v);
      }
      // per-wave B tile, stored transposed: Bt[col][k].
      // Each half-wave reads 16 consecutive couts for a fixed cin (coalesced).
      for (int i = 0; i < 16; ++i) {
        int kk  = i * 2 + half;
        int cin = k0 + kk;
        int co  = coutBase + mrow;
        float v = 0.f;
        if (cin < Cin && co < Cout) v = W[((size_t)s * Cin + cin) * Cout + co];
        Bt[wv][mrow][kk] = f2bf(v);
      }
      __syncthreads();

      // fragments per CDNA5 16-bit A(16x32)/B(32x16) layouts, b128 LDS loads
      FragBF fa0, fa1, fb;
      {
        const unsigned short* pb = &Bt[wv][mrow][half * 8];
        fb.q[0] = *(const uint4*)(pb);
        fb.q[1] = *(const uint4*)(pb + 16);
        const unsigned short* pa0 = &As[mrow][half * 8];
        fa0.q[0] = *(const uint4*)(pa0);
        fa0.q[1] = *(const uint4*)(pa0 + 16);
        const unsigned short* pa1 = &As[16 + mrow][half * 8];
        fa1.q[0] = *(const uint4*)(pa1);
        fa1.q[1] = *(const uint4*)(pa1 + 16);
      }
      acc0 = __builtin_amdgcn_wmma_f32_16x16x32_bf16(
          false, fa0.v, false, fb.v, (short)0, acc0, false, false);
      acc1 = __builtin_amdgcn_wmma_f32_16x16x32_bf16(
          false, fa1.v, false, fb.v, (short)0, acc1, false, false);
      __syncthreads();
    }
  }

  // epilogue: C/D layout — lanes 0-15: M=r, lanes 16-31: M=r+8; N = lane%16
  const int co = coutBase + mrow;
  if (co < Cout) {
    float bsv = bias ? bias[co] : 0.f;
    float inv = 0.f, bb = 0.f, mm = 0.f;
    if (mode == SC_BNELU || mode == SC_DISP) {
      inv = rsqrtf(vr[co] + BN_EPS) * g[co];
      bb  = be[co];
      mm  = mn[co];
    }
#pragma unroll
    for (int t = 0; t < 2; ++t) {
      const v8f& acc = t ? acc1 : acc0;
      const int mbase = m0 + t * 16 + half * 8;
#pragma unroll
      for (int r = 0; r < 8; ++r) {
        int node = mbase + r;
        float v = acc[r] + bsv;
        if (mode == SC_BNELU) {
          v = (v - mm) * inv + bb;
          v = v > 0.f ? v : (expf(v) - 1.f);
          out[(size_t)node * Cout + co] = v;
        } else if (mode == SC_DISP) {
          v = (v - mm) * inv + bb;
          v = 0.3f / (1.f + expf(-v));
          out[(size_t)node * Cout + co] = v;
        } else if (mode == SC_POOL) {
          atomicAdd(&out[(size_t)cluster[node] * Cout + co], 0.25f * v);
        } else {
          out[(size_t)node * Cout + co] = v;
        }
      }
    }
  }
}

// ---------------------------------------------------------------------------
// glue kernels
// ---------------------------------------------------------------------------
__global__ void nbr_init_kernel(int* nbr, int n) {
  int i = blockIdx.x * blockDim.x + threadIdx.x;
  if (i < n) nbr[i] = -1;
}
__global__ void nbr_scatter_kernel(const int* __restrict__ src,
                                   const int* __restrict__ dst,
                                   const int* __restrict__ sel,
                                   int* __restrict__ nbr, int E) {
  int e = blockIdx.x * blockDim.x + threadIdx.x;
  if (e < E) nbr[dst[e] * 9 + sel[e]] = src[e];
}
__global__ void zero_f32_kernel(float* p, long long n) {
  long long i = blockIdx.x * (long long)blockDim.x + threadIdx.x;
  if (i < n) p[i] = 0.f;
}
__global__ void set_u32_kernel(unsigned int* p, unsigned int v, long long n) {
  long long i = blockIdx.x * (long long)blockDim.x + threadIdx.x;
  if (i < n) p[i] = v;
}
__global__ void bn_elu_kernel(float* io, const float* g, const float* be,
                              const float* mn, const float* vr,
                              long long N, int C) {
  long long i = blockIdx.x * (long long)blockDim.x + threadIdx.x;
  if (i >= N * C) return;
  int c = (int)(i % C);
  float v = io[i];
  v = (v - mn[c]) * rsqrtf(vr[c] + BN_EPS) * g[c] + be[c];
  io[i] = v > 0.f ? v : (expf(v) - 1.f);
}
__global__ void add_bn_elu_kernel(const float* a, const float* b,
                                  const float* g, const float* be,
                                  const float* mn, const float* vr,
                                  float* o, long long N, int C) {
  long long i = blockIdx.x * (long long)blockDim.x + threadIdx.x;
  if (i >= N * C) return;
  int c = (int)(i % C);
  float v = a[i] + b[i];
  v = (v - mn[c]) * rsqrtf(vr[c] + BN_EPS) * g[c] + be[c];
  o[i] = v > 0.f ? v : (expf(v) - 1.f);
}
__global__ void pool_avg_kernel(const float* __restrict__ x,
                                const int* __restrict__ cl,
                                float* __restrict__ o, int Nf, int C) {
  long long i = blockIdx.x * (long long)blockDim.x + threadIdx.x;
  if (i >= (long long)Nf * C) return;
  int n = (int)(i / C), c = (int)(i % C);
  atomicAdd(&o[(size_t)cl[n] * C + c], 0.25f * x[i]);
}
__device__ __forceinline__ unsigned int encf(float f) {
  unsigned int u = __float_as_uint(f);
  return (u >> 31) ? ~u : (u | 0x80000000u);
}
__device__ __forceinline__ float decf(unsigned int e) {
  return (e & 0x80000000u) ? __uint_as_float(e & 0x7FFFFFFFu)
                           : __uint_as_float(~e);
}
__global__ void maxpool_kernel(const float* __restrict__ x,
                               const int* __restrict__ nbr,
                               const int* __restrict__ cl,
                               unsigned int* __restrict__ oenc,
                               int Nf, int C) {
  long long i = blockIdx.x * (long long)blockDim.x + threadIdx.x;
  if (i >= (long long)Nf * C) return;
  int n = (int)(i / C), c = (int)(i % C);
  float m = -INFINITY;
  for (int s = 0; s < 9; ++s) {
    int q = nbr[n * 9 + s];
    if (q >= 0) m = fmaxf(m, x[(size_t)q * C + c]);
  }
  atomicMax(&oenc[(size_t)cl[n] * C + c], encf(m));
}
__global__ void dec_u32_kernel(const unsigned int* e, float* o, long long n) {
  long long i = blockIdx.x * (long long)blockDim.x + threadIdx.x;
  if (i < n) o[i] = decf(e[i]);
}
__global__ void unpool_kernel(const float* __restrict__ xc,
                              const int* __restrict__ cl,
                              const int* __restrict__ nbr,
                              float* __restrict__ o, int Nf, int C) {
  long long i = blockIdx.x * (long long)blockDim.x + threadIdx.x;
  if (i >= (long long)Nf * C) return;
  int n = (int)(i / C), c = (int)(i % C);
  float s = 0.f; int deg = 0;
  for (int k = 0; k < 9; ++k) {
    int m = nbr[n * 9 + k];
    if (m >= 0) { s += xc[(size_t)cl[m] * C + c]; deg++; }
  }
  o[i] = s / (float)deg;
}
__global__ void copy_cols_kernel(const float* __restrict__ src, int Cs,
                                 float* __restrict__ dst, int Cd, int off,
                                 long long N) {
  long long i = blockIdx.x * (long long)blockDim.x + threadIdx.x;
  if (i >= N * Cs) return;
  long long n = i / Cs; int c = (int)(i % Cs);
  dst[n * Cd + off + c] = src[i];
}

// ---------------------------------------------------------------------------
// host orchestration
// ---------------------------------------------------------------------------
extern "C" void kernel_launch(void* const* d_in, const int* in_sizes, int n_in,
                              void* d_out, int out_size, void* d_ws, size_t ws_size,
                              hipStream_t stream) {
  (void)n_in; (void)out_size; (void)ws_size;
  int cur = 0;
  const float* x_in = (const float*)d_in[cur++];
  const int N0 = in_sizes[0] / 3;
  int Nl[7]; Nl[0] = N0; for (int i = 1; i < 7; ++i) Nl[i] = Nl[i - 1] / 4;

  auto Fp = [&]() { return (const float*)d_in[cur++]; };

  struct CBlk { const float *W, *b, *g, *be, *mn, *vr; int Cin, Cout; };
  struct Lin  { const float *W, *b; int Cin, Cout; };
  struct BNp  { const float *g, *be, *mn, *vr; };
  struct RBlk { CBlk c1, c2; Lin c3; BNp bn; };

  auto readCB = [&](int ci, int co) { CBlk c; c.W=Fp(); c.b=Fp(); c.g=Fp(); c.be=Fp(); c.mn=Fp(); c.vr=Fp(); c.Cin=ci; c.Cout=co; return c; };
  auto readLn = [&](int ci, int co) { Lin l; l.W=Fp(); l.b=Fp(); l.Cin=ci; l.Cout=co; return l; };
  auto readBN = [&]() { BNp b; b.g=Fp(); b.be=Fp(); b.mn=Fp(); b.vr=Fp(); return b; };
  auto readRB = [&](int ci, int co) { RBlk r; r.c1=readCB(ci,co); r.c2=readCB(co,co); r.c3=readLn(ci,co); r.bn=readBN(); return r; };

  CBlk conv1p = readCB(3, 64);
  RBlk dconv2 = readRB(64, 64),   conv2p = readRB(64, 64);
  RBlk dconv3 = readRB(64, 128),  conv3p = readRB(128, 128);
  RBlk dconv4 = readRB(128, 256), conv4p = readRB(256, 256);
  RBlk dconv5 = readRB(256, 512), conv5p = readRB(512, 512);
  CBlk upc6 = readCB(512, 512);
  CBlk ic6  = readCB(768, 512);
  CBlk upc5 = readCB(512, 256);
  CBlk ic5  = readCB(384, 256);
  CBlk upc4 = readCB(256, 128);
  CBlk ic4  = readCB(192, 128);
  CBlk dsp4 = readCB(128, 2);
  CBlk upc3 = readCB(128, 64);
  CBlk ic3  = readCB(130, 64);
  CBlk dsp3 = readCB(64, 2);
  CBlk upc2 = readCB(64, 32);
  CBlk ic2  = readCB(98, 32);
  CBlk dsp2 = readCB(32, 2);
  CBlk upc1 = readCB(32, 16);
  CBlk ic1  = readCB(18, 16);
  CBlk dsp1 = readCB(16, 2);

  const int* ei[7]; int E[7];
  for (int l = 0; l < 7; ++l) { ei[l] = (const int*)d_in[cur]; E[l] = in_sizes[cur] / 2; cur++; }
  const int* sel[7];
  for (int l = 0; l < 7; ++l) sel[l] = (const int*)d_in[cur++];
  const int* cl[6];
  for (int l = 0; l < 6; ++l) cl[l] = (const int*)d_in[cur++];

  char* base = (char*)d_ws; size_t off = 0;
  auto alloc = [&](size_t bytes) -> void* {
    void* p = base + off; off = (off + bytes + 255) & ~(size_t)255; return p;
  };
  auto gl = [](long long n) { return dim3((unsigned)((n + 255) / 256)); };

  // neighbor maps (inverse edge maps) per level
  int* nbr[7];
  for (int l = 0; l < 7; ++l) {
    nbr[l] = (int*)alloc((size_t)Nl[l] * 9 * sizeof(int));
    long long tot = (long long)Nl[l] * 9;
    nbr_init_kernel<<<gl(tot), 256, 0, stream>>>(nbr[l], (int)tot);
    nbr_scatter_kernel<<<gl(E[l]), 256, 0, stream>>>(ei[l], ei[l] + E[l], sel[l], nbr[l], E[l]);
  }

  // persistent activations
  float* x1  = (float*)alloc((size_t)Nl[1] * 64 * 4);
  float* xp1 = (float*)alloc((size_t)Nl[2] * 64 * 4);
  float* x2  = (float*)alloc((size_t)Nl[3] * 64 * 4);
  float* x3  = (float*)alloc((size_t)Nl[4] * 128 * 4);
  float* x4  = (float*)alloc((size_t)Nl[5] * 256 * 4);
  float* x5  = (float*)alloc((size_t)Nl[6] * 512 * 4);
  float* i6  = (float*)alloc((size_t)Nl[5] * 512 * 4);
  float* i5  = (float*)alloc((size_t)Nl[4] * 256 * 4);
  float* i4  = (float*)alloc((size_t)Nl[3] * 128 * 4);
  float* i3  = (float*)alloc((size_t)Nl[2] * 64 * 4);
  float* i2  = (float*)alloc((size_t)Nl[1] * 32 * 4);
  float* ud4 = (float*)alloc((size_t)Nl[2] * 2 * 4);
  float* ud3 = (float*)alloc((size_t)Nl[1] * 2 * 4);
  float* ud2 = (float*)alloc((size_t)Nl[0] * 2 * 4);
  const size_t arena = off;

  float* outF  = (float*)d_out;
  float* disp1 = outF;
  float* disp2 = disp1 + (size_t)Nl[0] * 2;
  float* disp3 = disp2 + (size_t)Nl[1] * 2;
  float* disp4 = disp3 + (size_t)Nl[2] * 2;

  auto sconv = [&](const float* xin, const int* nb, const float* W, const float* b,
                   const float* g, const float* be, const float* mn, const float* vr,
                   const int* clu, float* o, int N, int Ci, int Co, int ns, int mode) {
    dim3 grid((unsigned)(N / 32), (unsigned)((Co + 63) / 64));
    sconv_wmma_kernel<<<grid, 128, 0, stream>>>(xin, nb, W, b, g, be, mn, vr, clu, o, N, Ci, Co, ns, mode);
  };
  auto sconvCB = [&](const CBlk& p, const float* xin, const int* nb, float* o, int N,
                     int mode, const int* clu) {
    sconv(xin, nb, p.W, p.b, p.g, p.be, p.mn, p.vr, clu, o, N, p.Cin, p.Cout, 9, mode);
  };
  auto gemmLin = [&](const Lin& p, const float* xin, float* o, int N) {
    sconv(xin, nullptr, p.W, p.b, nullptr, nullptr, nullptr, nullptr, nullptr, o, N, p.Cin, p.Cout, 1, SC_PLAIN);
  };
  auto zero = [&](float* p, long long n) { zero_f32_kernel<<<gl(n), 256, 0, stream>>>(p, n); };
  auto poolavg = [&](const float* x, const int* c, float* o, int Nf, int C) {
    pool_avg_kernel<<<gl((long long)Nf * C), 256, 0, stream>>>(x, c, o, Nf, C);
  };
  auto addBnElu = [&](const float* a, const float* b, const BNp& bn, float* o, int N, int C) {
    add_bn_elu_kernel<<<gl((long long)N * C), 256, 0, stream>>>(a, b, bn.g, bn.be, bn.mn, bn.vr, o, N, C);
  };
  auto unpool = [&](const float* xc, const int* clu, const int* nb, float* o, int Nf, int C) {
    unpool_kernel<<<gl((long long)Nf * C), 256, 0, stream>>>(xc, clu, nb, o, Nf, C);
  };
  auto copyc = [&](const float* src, int Cs, float* dst, int Cd, int coff, int N) {
    copy_cols_kernel<<<gl((long long)N * Cs), 256, 0, stream>>>(src, Cs, dst, Cd, coff, N);
  };

  auto resconv = [&](const RBlk& p, const float* xin, int Nf, const int* nbrF,
                     bool strided, const int* clF, int Nc, const int* nbrC, float* o) {
    size_t save = off;
    const int Co = p.c1.Cout;
    float* t1 = (float*)alloc((size_t)Nf * Co * 4);
    sconvCB(p.c1, xin, nbrF, t1, Nf, SC_BNELU, nullptr);
    if (strided) {
      float* tp = (float*)alloc((size_t)Nc * Co * 4);
      zero(tp, (long long)Nc * Co);
      poolavg(t1, clF, tp, Nf, Co);
      float* xo = (float*)alloc((size_t)Nc * Co * 4);
      sconvCB(p.c2, tp, nbrC, xo, Nc, SC_BNELU, nullptr);
      float* scf = (float*)alloc((size_t)Nf * Co * 4);
      gemmLin(p.c3, xin, scf, Nf);
      float* scc = (float*)alloc((size_t)Nc * Co * 4);
      zero(scc, (long long)Nc * Co);
      poolavg(scf, clF, scc, Nf, Co);
      addBnElu(xo, scc, p.bn, o, Nc, Co);
    } else {
      float* xo = (float*)alloc((size_t)Nf * Co * 4);
      sconvCB(p.c2, t1, nbrF, xo, Nf, SC_BNELU, nullptr);
      float* sc = (float*)alloc((size_t)Nf * Co * 4);
      gemmLin(p.c3, xin, sc, Nf);
      addBnElu(xo, sc, p.bn, o, Nf, Co);
    }
    off = save;
  };

  // ---------------- encoder ----------------
  // conv1: selection conv at L0 with fused 2x2 avg-pool into L1, then BN+ELU
  off = arena;
  zero(x1, (long long)Nl[1] * 64);
  sconvCB(conv1p, x_in, nbr[0], x1, Nl[0], SC_POOL, cl[0]);
  bn_elu_kernel<<<gl((long long)Nl[1] * 64), 256, 0, stream>>>(
      x1, conv1p.g, conv1p.be, conv1p.mn, conv1p.vr, Nl[1], 64);

  // max pool: 9-neighborhood max then 2x2 max (encoded-u32 atomicMax)
  off = arena;
  {
    unsigned int* enc = (unsigned int*)alloc((size_t)Nl[2] * 64 * 4);
    set_u32_kernel<<<gl((long long)Nl[2] * 64), 256, 0, stream>>>(enc, 0u, (long long)Nl[2] * 64);
    maxpool_kernel<<<gl((long long)Nl[1] * 64), 256, 0, stream>>>(x1, nbr[1], cl[1], enc, Nl[1], 64);
    dec_u32_kernel<<<gl((long long)Nl[2] * 64), 256, 0, stream>>>(enc, xp1, (long long)Nl[2] * 64);
  }

  off = arena; resconv(dconv2, xp1, Nl[2], nbr[2], true,  cl[2], Nl[3], nbr[3], x2);
  off = arena; resconv(conv2p, x2,  Nl[3], nbr[3], false, nullptr, 0, nullptr, x2);
  off = arena; resconv(dconv3, x2,  Nl[3], nbr[3], true,  cl[3], Nl[4], nbr[4], x3);
  off = arena; resconv(conv3p, x3,  Nl[4], nbr[4], false, nullptr, 0, nullptr, x3);
  off = arena; resconv(dconv4, x3,  Nl[4], nbr[4], true,  cl[4], Nl[5], nbr[5], x4);
  off = arena; resconv(conv4p, x4,  Nl[5], nbr[5], false, nullptr, 0, nullptr, x4);
  off = arena; resconv(dconv5, x4,  Nl[5], nbr[5], true,  cl[5], Nl[6], nbr[6], x5);
  off = arena; resconv(conv5p, x5,  Nl[6], nbr[6], false, nullptr, 0, nullptr, x5);

  // ---------------- decoder ----------------
  off = arena;
  { // up6 + i6 (L5)
    float* t   = (float*)alloc((size_t)Nl[5] * 512 * 4);
    unpool(x5, cl[5], nbr[5], t, Nl[5], 512);
    float* up6 = (float*)alloc((size_t)Nl[5] * 512 * 4);
    sconvCB(upc6, t, nbr[5], up6, Nl[5], SC_BNELU, nullptr);
    float* cat = (float*)alloc((size_t)Nl[5] * 768 * 4);
    copyc(up6, 512, cat, 768, 0, Nl[5]);
    copyc(x4, 256, cat, 768, 512, Nl[5]);
    sconvCB(ic6, cat, nbr[5], i6, Nl[5], SC_BNELU, nullptr);
  }
  off = arena;
  { // up5 + i5 (L4)
    float* t   = (float*)alloc((size_t)Nl[4] * 512 * 4);
    unpool(i6, cl[4], nbr[4], t, Nl[4], 512);
    float* up5 = (float*)alloc((size_t)Nl[4] * 256 * 4);
    sconvCB(upc5, t, nbr[4], up5, Nl[4], SC_BNELU, nullptr);
    float* cat = (float*)alloc((size_t)Nl[4] * 384 * 4);
    copyc(up5, 256, cat, 384, 0, Nl[4]);
    copyc(x3, 128, cat, 384, 256, Nl[4]);
    sconvCB(ic5, cat, nbr[4], i5, Nl[4], SC_BNELU, nullptr);
  }
  off = arena;
  { // up4 + i4 (L3)
    float* t   = (float*)alloc((size_t)Nl[3] * 256 * 4);
    unpool(i5, cl[3], nbr[3], t, Nl[3], 256);
    float* up4 = (float*)alloc((size_t)Nl[3] * 128 * 4);
    sconvCB(upc4, t, nbr[3], up4, Nl[3], SC_BNELU, nullptr);
    float* cat = (float*)alloc((size_t)Nl[3] * 192 * 4);
    copyc(up4, 128, cat, 192, 0, Nl[3]);
    copyc(x2, 64, cat, 192, 128, Nl[3]);
    sconvCB(ic4, cat, nbr[3], i4, Nl[3], SC_BNELU, nullptr);
  }
  off = arena;
  sconvCB(dsp4, i4, nbr[3], disp4, Nl[3], SC_DISP, nullptr);
  unpool(disp4, cl[2], nbr[2], ud4, Nl[2], 2);

  off = arena;
  { // up3 + i3 (L2)
    float* t   = (float*)alloc((size_t)Nl[2] * 128 * 4);
    unpool(i4, cl[2], nbr[2], t, Nl[2], 128);
    float* up3 = (float*)alloc((size_t)Nl[2] * 64 * 4);
    sconvCB(upc3, t, nbr[2], up3, Nl[2], SC_BNELU, nullptr);
    float* cat = (float*)alloc((size_t)Nl[2] * 130 * 4);
    copyc(up3, 64, cat, 130, 0, Nl[2]);
    copyc(xp1, 64, cat, 130, 64, Nl[2]);
    copyc(ud4, 2, cat, 130, 128, Nl[2]);
    sconvCB(ic3, cat, nbr[2], i3, Nl[2], SC_BNELU, nullptr);
  }
  off = arena;
  sconvCB(dsp3, i3, nbr[2], disp3, Nl[2], SC_DISP, nullptr);
  unpool(disp3, cl[1], nbr[1], ud3, Nl[1], 2);

  off = arena;
  { // up2 + i2 (L1)
    float* t   = (float*)alloc((size_t)Nl[1] * 64 * 4);
    unpool(i3, cl[1], nbr[1], t, Nl[1], 64);
    float* up2 = (float*)alloc((size_t)Nl[1] * 32 * 4);
    sconvCB(upc2, t, nbr[1], up2, Nl[1], SC_BNELU, nullptr);
    float* cat = (float*)alloc((size_t)Nl[1] * 98 * 4);
    copyc(up2, 32, cat, 98, 0, Nl[1]);
    copyc(x1, 64, cat, 98, 32, Nl[1]);
    copyc(ud3, 2, cat, 98, 96, Nl[1]);
    sconvCB(ic2, cat, nbr[1], i2, Nl[1], SC_BNELU, nullptr);
  }
  off = arena;
  sconvCB(dsp2, i2, nbr[1], disp2, Nl[1], SC_DISP, nullptr);
  unpool(disp2, cl[0], nbr[0], ud2, Nl[0], 2);

  off = arena;
  { // up1 + i1 + disp1 (L0)
    float* t   = (float*)alloc((size_t)Nl[0] * 32 * 4);
    unpool(i2, cl[0], nbr[0], t, Nl[0], 32);
    float* up1 = (float*)alloc((size_t)Nl[0] * 16 * 4);
    sconvCB(upc1, t, nbr[0], up1, Nl[0], SC_BNELU, nullptr);
    float* cat = (float*)alloc((size_t)Nl[0] * 18 * 4);
    copyc(up1, 16, cat, 18, 0, Nl[0]);
    copyc(ud2, 2, cat, 18, 16, Nl[0]);
    float* i1 = (float*)alloc((size_t)Nl[0] * 16 * 4);
    sconvCB(ic1, cat, nbr[0], i1, Nl[0], SC_BNELU, nullptr);
    sconvCB(dsp1, i1, nbr[0], disp1, Nl[0], SC_DISP, nullptr);
  }
}
